// GPT2WithXLSTM_5944234738091
// MI455X (gfx1250) — matrix operand order repaired and checked by reading, not compiled, not run on hardware
//
#include <hip/hip_runtime.h>
#include <hip/hip_bf16.h>
#include <math.h>

typedef __attribute__((ext_vector_type(16))) __bf16 v16bf;
typedef __attribute__((ext_vector_type(8)))  __bf16 v8bf;
typedef __attribute__((ext_vector_type(8)))  float  v8f;

#define B_ 4
#define S_ 1024
#define D_ 768
#define H_ 256
#define G_ 1024            // 4*H
#define V_ 50257
#define M_ 4096            // B*S

// -------- workspace layout (bytes, 256-aligned) --------
#define OFF_XPROJ   0u            // f32 [4096,1024]  = 16777216
#define OFF_ABF     16777216u     // bf16 [4096,768]  =  6291456
#define OFF_WIHBF   23068672u     // bf16 [1024,768]  =  1572864
#define OFF_WHHBF   24641536u     // bf16 [1024,256]  =   524288
#define OFF_WOUTBF  25165824u     // bf16 [50257,256] = 25731584
#define OFF_HSBF    50897408u     // bf16 [4096,256]  =  2097152
#define OFF_BIAS2   52994560u     // f32  [1024]      =     4096

__device__ __forceinline__ float sigmoidf_(float x) {
  return 1.0f / (1.0f + __expf(-x));
}

__device__ __forceinline__ v8f wmma_bf16(v16bf a, v16bf b, v8f c) {
  return __builtin_amdgcn_wmma_f32_16x16x32_bf16(false, a, false, b, (short)0, c,
                                                 false, false);
}

// Build A fragment (16x32 bf16). Per-lane: row = lane&15; K-halves by lane>>4.
__device__ __forceinline__ v16bf make_a_frag(const __bf16* ap) {
  v8bf lo = *(const v8bf*)(ap);
  v8bf hi = *(const v8bf*)(ap + 8);
  v16bf a;
#pragma unroll
  for (int i = 0; i < 8; ++i) { a[i] = lo[i]; a[8 + i] = hi[i]; }
  return a;
}

// ---------------- conversion kernels ----------------
__global__ void cvt_f32_bf16(const float* __restrict__ in, __bf16* __restrict__ out, int n) {
  int i = blockIdx.x * blockDim.x + threadIdx.x;
  if (i < n) out[i] = (__bf16)in[i];
}

__global__ void bias_sum(const float* __restrict__ a, const float* __restrict__ b,
                         float* __restrict__ out, int n) {
  int i = blockIdx.x * blockDim.x + threadIdx.x;
  if (i < n) out[i] = a[i] + b[i];
}

// ---------------- GEMM1: x_proj = X(bf16) * W_ih^T + (b_ih+b_hh) ----------------
// A: [4096,768] bf16. W: [1024,768] bf16 (N,K). C: [4096,1024] f32.
// Each wave: 2 M-tiles x 4 N-tiles (register blocked). 8 waves/block.
// Grid: (4096/32) * (1024/64) = 128*16 = 2048 waves = 256 blocks.
__global__ __launch_bounds__(256) void gemm_xproj(const __bf16* __restrict__ A,
                                                  const __bf16* __restrict__ W,
                                                  const float* __restrict__ bias,
                                                  float* __restrict__ C) {
  const int lane = threadIdx.x & 31;
  const int gw   = blockIdx.x * 8 + (threadIdx.x >> 5);
  const int mb   = gw >> 4;            // 128 M-blocks of 32 rows
  const int nb   = gw & 15;            // 16 N-blocks of 64 cols
  const int m0 = mb * 32, n0 = nb * 64;
  const int row = lane & 15;
  const int ah  = lane >> 4;

  const __bf16* ap0 = A + (size_t)(m0 + row) * D_ + ah * 16;
  const __bf16* ap1 = ap0 + 16 * D_;
  const __bf16* bp[4];
#pragma unroll
  for (int u = 0; u < 4; ++u)
    bp[u] = W + (size_t)(n0 + u * 16 + row) * D_ + ah * 16;

  v8f acc[2][4] = {};
  for (int k0 = 0; k0 < D_; k0 += 32) {
    v16bf a0 = make_a_frag(ap0 + k0);
    v16bf a1 = make_a_frag(ap1 + k0);
    __builtin_prefetch(bp[0] + k0 + 128, 0, 3);
#pragma unroll
    for (int u = 0; u < 4; ++u) {
      v16bf b = *(const v16bf*)(bp[u] + k0);
      acc[0][u] = wmma_bf16(a0, b, acc[0][u]);
      acc[1][u] = wmma_bf16(a1, b, acc[1][u]);
    }
  }
#pragma unroll
  for (int u = 0; u < 4; ++u) {
    const int ccol = n0 + u * 16 + row;
    const float bv = bias[ccol];
#pragma unroll
    for (int m = 0; m < 2; ++m) {
      const int crow = m0 + m * 16 + ah * 8;
#pragma unroll
      for (int j = 0; j < 8; ++j)
        C[(size_t)(crow + j) * G_ + ccol] = acc[m][u][j] + bv;
    }
  }
}

// ---------------- LSTM scan: persistent single workgroup (32 waves) ----------------
// xproj: f32 [B,S,G] ; Whh: bf16 [G,H] (N,K) ; hsbf out: bf16 [B*S, H]
__global__ __launch_bounds__(1024) void lstm_scan(const float* __restrict__ xproj,
                                                  const __bf16* __restrict__ Whh,
                                                  __bf16* __restrict__ hsbf) {
  __shared__ __bf16 h_bf[16 * H_];   // A operand, rows 4..15 stay zero
  __shared__ float  c_st[B_ * H_];
  __shared__ float  gates[B_ * G_];

  const int tid = threadIdx.x;
  for (int i = tid; i < 16 * H_; i += 1024) h_bf[i] = (__bf16)0.0f;
  for (int i = tid; i < B_ * H_; i += 1024) c_st[i] = 0.0f;
  __syncthreads();

  const int lane = tid & 31;
  const int wv   = tid >> 5;          // 0..31
  const int row  = lane & 15;
  const int ah   = lane >> 4;
  const __bf16* ap = h_bf + row * H_ + ah * 16;
  const int eb = tid >> 8;            // elementwise batch index
  const int eh = tid & 255;           // elementwise hidden index

  for (int t = 0; t < S_; ++t) {
    // A fragments for full K=256 (shared across this wave's two N tiles)
    v16bf afr[8];
#pragma unroll
    for (int kk = 0; kk < 8; ++kk) afr[kk] = make_a_frag(ap + kk * 32);

#pragma unroll
    for (int sub = 0; sub < 2; ++sub) {
      const int n0 = (wv * 2 + sub) * 16;
      const __bf16* bp = Whh + (size_t)(n0 + row) * H_ + ah * 16;
      v8f acc = {};
#pragma unroll
      for (int kk = 0; kk < 8; ++kk) {
        v16bf b = *(const v16bf*)(bp + kk * 32);
        acc = wmma_bf16(afr[kk], b, acc);
      }
      if (ah == 0) {                  // lanes 0..15 hold rows 0..7; rows 0..3 valid
        const int ccol = n0 + row;
#pragma unroll
        for (int j = 0; j < B_; ++j)
          gates[j * G_ + ccol] = acc[j] + xproj[((size_t)j * S_ + t) * G_ + ccol];
      }
    }
    __syncthreads();

    // elementwise gate math; one thread per (b,h)
    {
      float gi = gates[eb * G_ + eh];
      float gf = gates[eb * G_ + H_ + eh];
      float gg = gates[eb * G_ + 2 * H_ + eh];
      float go = gates[eb * G_ + 3 * H_ + eh];
      float c  = c_st[eb * H_ + eh];
      c = sigmoidf_(gf) * c + sigmoidf_(gi) * tanhf(gg);
      float hv = sigmoidf_(go) * tanhf(c);
      c_st[eb * H_ + eh] = c;
      h_bf[eb * H_ + eh] = (__bf16)hv;
      hsbf[((size_t)eb * S_ + t) * H_ + eh] = (__bf16)hv;
    }
    __syncthreads();
  }
}

// ---------------- GEMM2: logits = hs(bf16) * W_out^T + b_out ----------------
// Hs: [4096,256] bf16 ; W: [V,256] bf16 ; C: [4096,V] f32.
// Each wave: 2 M-tiles x 4 N-tiles. Grid: (4096/32) * ceil(V/64) = 128*786 waves.
#define NB_OUT 786   // ceil(V/64); 786*64 = 50304 (edge tiles clamped)
__global__ __launch_bounds__(256) void gemm_logits(const __bf16* __restrict__ Hs,
                                                   const __bf16* __restrict__ W,
                                                   const float* __restrict__ bias,
                                                   float* __restrict__ C) {
  const int lane = threadIdx.x & 31;
  const int gw   = blockIdx.x * 8 + (threadIdx.x >> 5);
  const int mb   = gw / NB_OUT;
  const int nb   = gw - mb * NB_OUT;
  const int m0 = mb * 32, n0 = nb * 64;
  const int row = lane & 15;
  const int ah  = lane >> 4;

  const __bf16* ap0 = Hs + (size_t)(m0 + row) * H_ + ah * 16;
  const __bf16* ap1 = ap0 + 16 * H_;
  const __bf16* bp[4];
#pragma unroll
  for (int u = 0; u < 4; ++u) {
    int ncl = n0 + u * 16 + row;
    if (ncl > V_ - 1) ncl = V_ - 1;   // clamp edge-tile loads in-bounds
    bp[u] = W + (size_t)ncl * H_ + ah * 16;
  }

  v8f acc[2][4] = {};
#pragma unroll
  for (int k0 = 0; k0 < H_; k0 += 32) {
    v16bf a0 = make_a_frag(ap0 + k0);
    v16bf a1 = make_a_frag(ap1 + k0);
#pragma unroll
    for (int u = 0; u < 4; ++u) {
      v16bf b = *(const v16bf*)(bp[u] + k0);
      acc[0][u] = wmma_bf16(a0, b, acc[0][u]);
      acc[1][u] = wmma_bf16(a1, b, acc[1][u]);
    }
  }
#pragma unroll
  for (int u = 0; u < 4; ++u) {
    const int ccol = n0 + u * 16 + row;
    if (ccol < V_) {
      const float bv = bias[ccol];
#pragma unroll
      for (int m = 0; m < 2; ++m) {
        const int crow = m0 + m * 16 + ah * 8;
#pragma unroll
        for (int j = 0; j < 8; ++j)
          C[(size_t)(crow + j) * V_ + ccol] = acc[m][u][j] + bv;
      }
    }
  }
}

extern "C" void kernel_launch(void* const* d_in, const int* in_sizes, int n_in,
                              void* d_out, int out_size, void* d_ws, size_t ws_size,
                              hipStream_t stream) {
  (void)in_sizes; (void)n_in; (void)out_size; (void)ws_size;
  const float* hid   = (const float*)d_in[0];  // [B,S,D]
  const float* W_ih  = (const float*)d_in[1];  // [4H,D]
  const float* W_hh  = (const float*)d_in[2];  // [4H,H]
  const float* b_ih  = (const float*)d_in[3];  // [4H]
  const float* b_hh  = (const float*)d_in[4];  // [4H]
  const float* W_out = (const float*)d_in[5];  // [V,H]
  const float* b_out = (const float*)d_in[6];  // [V]
  float* logits = (float*)d_out;

  char* ws = (char*)d_ws;
  float*  xproj  = (float*)(ws + OFF_XPROJ);
  __bf16* abf    = (__bf16*)(ws + OFF_ABF);
  __bf16* wihbf  = (__bf16*)(ws + OFF_WIHBF);
  __bf16* whhbf  = (__bf16*)(ws + OFF_WHHBF);
  __bf16* woutbf = (__bf16*)(ws + OFF_WOUTBF);
  __bf16* hsbf   = (__bf16*)(ws + OFF_HSBF);
  float*  bias2  = (float*)(ws + OFF_BIAS2);

  // precision-downconvert operands to bf16 in workspace
  cvt_f32_bf16<<<(M_ * D_) / 256, 256, 0, stream>>>(hid, abf, M_ * D_);
  cvt_f32_bf16<<<(G_ * D_) / 256, 256, 0, stream>>>(W_ih, wihbf, G_ * D_);
  cvt_f32_bf16<<<(G_ * H_) / 256, 256, 0, stream>>>(W_hh, whhbf, G_ * H_);
  cvt_f32_bf16<<<(V_ * H_ + 255) / 256, 256, 0, stream>>>(W_out, woutbf, V_ * H_);
  bias_sum<<<(G_ + 255) / 256, 256, 0, stream>>>(b_ih, b_hh, bias2, G_);

  // phase 1: input projection, all timesteps in one WMMA GEMM (2x4 reg-blocked)
  gemm_xproj<<<(128 * 16) / 8, 256, 0, stream>>>(abf, wihbf, bias2, xproj);

  // phase 2: sequential LSTM scan, persistent single workgroup
  lstm_scan<<<1, 1024, 0, stream>>>(xproj, whhbf, hsbf);

  // phase 3: vocabulary projection (HBM-store-bound; 2x4 blocking keeps L2 reads
  // ~4.8 GB, under the 823 MB @ 23.3 TB/s store roofline's time budget)
  gemm_logits<<<(128 * NB_OUT) / 8, 256, 0, stream>>>(hsbf, woutbf, b_out, logits);
}
